// Head_10230612099295
// MI455X (gfx1250) — matrix operand order repaired and checked by reading, not compiled, not run on hardware
//
#include <hip/hip_runtime.h>
#include <hip/hip_bf16.h>

// CDNA5 / gfx1250 fused causal attention:
//   kernel 1: Q,K,V = x @ W{q,k,v}  (bf16 out, 1/sqrt(C) folded into Q)
//   kernel 2: flash attention, online softmax, f32 WMMA accumulation
// Data path: global_load_async_to_lds_b128 (ASYNCcnt) for raw tiles,
//            vectorized ds_load_b128 fragment feeds, v_wmma_f32_16x16x32_bf16.
// Workspace (d_ws): Q | K | V bf16, 3 * 16384*64*2 = 6 MB.

typedef __attribute__((ext_vector_type(16))) __bf16 v16bf;
typedef __attribute__((ext_vector_type(8)))  float  v8f;

constexpr int Bn = 8, Tn = 2048, Cn = 1024, Hn = 64;
constexpr int Nrows = Bn * Tn;          // 16384 flattened rows

union U16 { uint4 q[2]; v16bf v; };     // 32B: one bf16 A/B fragment per lane
union UF8 { float4 f[2]; float s[8]; };
union UV8 { uint4 q; __bf16 e[8]; };

__device__ __forceinline__ v8f wmma_bf16(v16bf a, v16bf b, v8f c) {
    // D = A(16x32 bf16) * B(32x16 bf16) + C(16x16 f32)
    return __builtin_amdgcn_wmma_f32_16x16x32_bf16(
        false, a, false, b, (short)0, c, false, false);
}

// ISA 10.2: flat addresses in the LDS aperture use addr[31:0] as the LDS
// byte address -> truncating a generic pointer to __shared__ gives the
// LDS-relative address the async ops want in their VDST VGPR.
__device__ __forceinline__ unsigned lds_addr(const void* p) {
    return (unsigned)(unsigned long long)p;
}
// Async DMA: global -> LDS, 16 bytes, tracked by ASYNCcnt.
__device__ __forceinline__ void async_b128(unsigned lds, const void* g) {
    asm volatile("global_load_async_to_lds_b128 %0, %1, off"
                 :: "v"(lds), "v"((unsigned long long)g) : "memory");
}
__device__ __forceinline__ void wait_async0() {
    asm volatile("s_wait_asynccnt 0x0" ::: "memory");
}

// ---------------------------------------------------------------------------
// Kernel 1: QKV projection. grid.x = 3 * (Nrows/16), block = 128 (4 waves).
// Wave w owns cols [w*16, w*16+16) of H=64; all waves share the x A-tile.
// ---------------------------------------------------------------------------
__global__ void __launch_bounds__(128)
qkv_proj_kernel(const float* __restrict__ x,
                const float* __restrict__ Wq,
                const float* __restrict__ Wk,
                const float* __restrict__ Wv,
                __bf16* __restrict__ Qb,
                __bf16* __restrict__ Kb,
                __bf16* __restrict__ Vb)
{
    __shared__ float  xs[16][36];       // 16x32 f32 chunk (row 144B, 16B-mult)
    __shared__ __bf16 Wt[64][40];       // W chunk transposed [col][k] (row 80B)

    const int m    = blockIdx.x / (Nrows / 16); // 0=Q, 1=K, 2=V
    const int rt   = blockIdx.x % (Nrows / 16);
    const float* W = (m == 0) ? Wq : (m == 1) ? Wk : Wv;
    __bf16* obuf   = (m == 0) ? Qb : (m == 1) ? Kb : Vb;
    const float scale = (m == 0) ? 0.03125f : 1.0f;   // C^-0.5 = 1/32 into Q

    const int tid  = threadIdx.x;
    const int wave = tid >> 5;
    const int lane = tid & 31;
    const int half = lane >> 4;
    const int ln   = lane & 15;

    v8f acc = {};
    for (int k0 = 0; k0 < Cn; k0 += 32) {
        __syncthreads();
        // async-DMA the 16x32 f32 x chunk: one b128 per thread (exactly 2KB)
        {
            int idx = tid * 4;
            int r = idx >> 5, c = idx & 31;
            async_b128(lds_addr(&xs[r][c]),
                       &x[(size_t)(rt * 16 + r) * Cn + k0 + c]);
        }
        // stage W chunk transposed (f32 -> bf16), 16 elements per thread
        {
            int r = tid >> 2;               // K row 0..31
            int c = (tid & 3) * 16;         // H col base
            const float4* wp = (const float4*)&W[(size_t)(k0 + r) * Hn + c];
            #pragma unroll
            for (int v4 = 0; v4 < 4; ++v4) {
                float4 w = wp[v4];
                Wt[c + v4 * 4 + 0][r] = (__bf16)w.x;
                Wt[c + v4 * 4 + 1][r] = (__bf16)w.y;
                Wt[c + v4 * 4 + 2][r] = (__bf16)w.z;
                Wt[c + v4 * 4 + 3][r] = (__bf16)w.w;
            }
        }
        wait_async0();
        __syncthreads();

        // A fragment: row ln, K runs [half*8,+8) and [16+half*8,+8), vectorized
        UF8 xa0, xa1;
        const float4* xp = (const float4*)&xs[ln][0];
        xa0.f[0] = xp[half * 2];     xa0.f[1] = xp[half * 2 + 1];
        xa1.f[0] = xp[4 + half * 2]; xa1.f[1] = xp[5 + half * 2];
        v16bf a;
        #pragma unroll
        for (int i = 0; i < 8; ++i) {
            a[i]     = (__bf16)xa0.s[i];
            a[8 + i] = (__bf16)xa1.s[i];
        }
        // B fragment: col ln+wave*16, K runs [half*8,+8) and [16+half*8,+8) in Wt
        U16 ub;
        const uint4* bp = (const uint4*)&Wt[wave * 16 + ln][0];
        ub.q[0] = bp[half]; ub.q[1] = bp[2 + half];

        acc = wmma_bf16(a, ub.v, acc);
    }

    #pragma unroll
    for (int j = 0; j < 8; ++j) {           // C/D: VGPR j -> row half*8+j
        int row = rt * 16 + half * 8 + j;
        obuf[(size_t)row * Hn + wave * 16 + ln] = (__bf16)(acc[j] * scale);
    }
}

// ---------------------------------------------------------------------------
// Kernel 2: flash attention. grid.x = B * (T/64), block = 128 (4 waves).
// Wave w owns 16 query rows; 32-key K/V chunks staged in LDS (K async DMA,
// V transposed so the P@V B-fragment is a contiguous ds_load_b128 pair).
// ---------------------------------------------------------------------------
__global__ void __launch_bounds__(128)
flash_attn_kernel(const __bf16* __restrict__ Qb,
                  const __bf16* __restrict__ Kb,
                  const __bf16* __restrict__ Vb,
                  float* __restrict__ out)
{
    __shared__ __bf16 Ks[32][72];           // K chunk row-major (row 144B)
    __shared__ __bf16 Vt[64][40];           // V chunk transposed [h][key]
    __shared__ __bf16 Ps[4][16][40];        // per-wave P tile (16x32)

    const int b    = blockIdx.x / (Tn / 64);
    const int qt   = blockIdx.x % (Tn / 64);
    const int tid  = threadIdx.x;
    const int wave = tid >> 5;
    const int lane = tid & 31;
    const int half = lane >> 4;
    const int ln   = lane & 15;

    const size_t base = (size_t)b * Tn;
    const int qrow0 = qt * 64 + wave * 16;

    // Q fragments (scale folded in): 16B vector loads straight from global
    v16bf qa0, qa1;
    {
        U16 u0, u1;
        const uint4* qp = (const uint4*)&Qb[(base + qrow0 + ln) * Hn];
        u0.q[0] = qp[2 * half];     u0.q[1] = qp[2 * half + 1];
        u1.q[0] = qp[4 + 2 * half]; u1.q[1] = qp[5 + 2 * half];
        qa0 = u0.v; qa1 = u1.v;
    }

    v8f acc[4] = {};                        // out tile: 16 rows x 64 H
    float mrow[8], lrow[8];
    #pragma unroll
    for (int j = 0; j < 8; ++j) { mrow[j] = -1e30f; lrow[j] = 0.0f; }

    const int nkc = (qt + 1) * 2;           // 32-key chunks: keys < (qt+1)*64
    for (int kc = 0; kc < nkc; ++kc) {
        const int kbase = kc * 32;
        __syncthreads();
        {
            int r = tid >> 2;               // key row 0..31
            int c = (tid & 3) * 16;         // h base 0,16,32,48
            // K chunk: async DMA, two b128 segments per thread
            const __bf16* kg = &Kb[(base + kbase + r) * Hn + c];
            unsigned kl = lds_addr(&Ks[r][c]);
            async_b128(kl,      kg);
            async_b128(kl + 16, kg + 8);
            // V chunk: load 16 bf16, store transposed into Vt
            const uint4* vp = (const uint4*)&Vb[(base + kbase + r) * Hn + c];
            UV8 v0, v1; v0.q = vp[0]; v1.q = vp[1];
            #pragma unroll
            for (int e = 0; e < 8; ++e) {
                Vt[c + e][r]     = v0.e[e];
                Vt[c + 8 + e][r] = v1.e[e];
            }
            if (kc + 1 < nkc)               // global_prefetch next K chunk
                __builtin_prefetch(&Kb[(base + kbase + 32 + r) * Hn + c], 0, 1);
        }
        wait_async0();
        __syncthreads();

        // S = q @ k^T : two 16x16 key tiles, K-dim = H=64 in 2 WMMAs each
        v8f s0 = {}, s1 = {};
        {
            U16 b0, b1;
            const uint4* p0 = (const uint4*)&Ks[ln][half * 16];
            const uint4* p1 = (const uint4*)&Ks[ln][32 + half * 16];
            b0.q[0] = p0[0]; b0.q[1] = p0[1];
            b1.q[0] = p1[0]; b1.q[1] = p1[1];
            s0 = wmma_bf16(qa0, b0.v, s0);
            s0 = wmma_bf16(qa1, b1.v, s0);
            const uint4* p2 = (const uint4*)&Ks[16 + ln][half * 16];
            const uint4* p3 = (const uint4*)&Ks[16 + ln][32 + half * 16];
            b0.q[0] = p2[0]; b0.q[1] = p2[1];
            b1.q[0] = p3[0]; b1.q[1] = p3[1];
            s1 = wmma_bf16(qa0, b0.v, s1);
            s1 = wmma_bf16(qa1, b1.v, s1);
        }

        // causal mask + online softmax (row spread over 16 lanes of a half)
        #pragma unroll
        for (int j = 0; j < 8; ++j) {
            int qr  = qrow0 + half * 8 + j;
            float v0 = (kbase + ln      <= qr) ? s0[j] : -1e30f;
            float v1 = (kbase + 16 + ln <= qr) ? s1[j] : -1e30f;
            float cm = fmaxf(v0, v1);
            cm = fmaxf(cm, __shfl_xor(cm, 1));
            cm = fmaxf(cm, __shfl_xor(cm, 2));
            cm = fmaxf(cm, __shfl_xor(cm, 4));
            cm = fmaxf(cm, __shfl_xor(cm, 8));
            float mnew = fmaxf(mrow[j], cm);
            float cf   = __expf(mrow[j] - mnew);
            float e0   = __expf(v0 - mnew);
            float e1   = __expf(v1 - mnew);
            float rs   = e0 + e1;
            rs += __shfl_xor(rs, 1);
            rs += __shfl_xor(rs, 2);
            rs += __shfl_xor(rs, 4);
            rs += __shfl_xor(rs, 8);
            lrow[j] = lrow[j] * cf + rs;
            mrow[j] = mnew;
            acc[0][j] *= cf; acc[1][j] *= cf; acc[2][j] *= cf; acc[3][j] *= cf;
            Ps[wave][half * 8 + j][ln]      = (__bf16)e0;
            Ps[wave][half * 8 + j][16 + ln] = (__bf16)e1;
        }
        __syncthreads();                    // Ps visible; Ks reads done

        // P (16x32) @ V (32x64): P as A fragment (two 16B runs per lane)
        U16 up;
        const uint4* pp = (const uint4*)&Ps[wave][ln][0];
        up.q[0] = pp[2 * half]; up.q[1] = pp[2 * half + 1];
        #pragma unroll
        for (int ht = 0; ht < 4; ++ht) {
            U16 ubv;
            const uint4* vq = (const uint4*)&Vt[ht * 16 + ln][half * 16];
            ubv.q[0] = vq[0]; ubv.q[1] = vq[1];
            acc[ht] = wmma_bf16(up.v, ubv.v, acc[ht]);
        }
    }

    #pragma unroll
    for (int j = 0; j < 8; ++j) {
        float inv = 1.0f / lrow[j];
        size_t row = base + qrow0 + half * 8 + j;
        out[row * Hn + ln]      = acc[0][j] * inv;
        out[row * Hn + 16 + ln] = acc[1][j] * inv;
        out[row * Hn + 32 + ln] = acc[2][j] * inv;
        out[row * Hn + 48 + ln] = acc[3][j] * inv;
    }
}

// ---------------------------------------------------------------------------
extern "C" void kernel_launch(void* const* d_in, const int* in_sizes, int n_in,
                              void* d_out, int out_size, void* d_ws, size_t ws_size,
                              hipStream_t stream) {
    (void)in_sizes; (void)n_in; (void)out_size; (void)ws_size;
    const float* x  = (const float*)d_in[0];
    const float* Wk = (const float*)d_in[1];
    const float* Wq = (const float*)d_in[2];
    const float* Wv = (const float*)d_in[3];
    float* out = (float*)d_out;

    __bf16* Qb = (__bf16*)d_ws;                       // 6 MB of d_ws used
    __bf16* Kb = Qb + (size_t)Nrows * Hn;
    __bf16* Vb = Kb + (size_t)Nrows * Hn;

    qkv_proj_kernel<<<3 * (Nrows / 16), 128, 0, stream>>>(x, Wq, Wk, Wv, Qb, Kb, Vb);
    flash_attn_kernel<<<Bn * (Tn / 64), 128, 0, stream>>>(Qb, Kb, Vb, out);
}
